// MultiHeadSelfAttention_42554535968869
// MI455X (gfx1250) — matrix operand order repaired and checked by reading, not compiled, not run on hardware
//
#include <hip/hip_runtime.h>
#include <hip/hip_bf16.h>

// ---------------------------------------------------------------------------
// MultiHeadSelfAttention for MI455X (gfx1250): bf16 WMMA everywhere,
// async-to-LDS double-buffered GEMM pipeline.
// B=2, T=2048, D=1024, H=16, dh=64.
// ---------------------------------------------------------------------------

typedef __attribute__((ext_vector_type(16))) __bf16 v16bf;
typedef __attribute__((ext_vector_type(4)))  __bf16 v4bf;
typedef __attribute__((ext_vector_type(8)))  float  v8f;
typedef __attribute__((ext_vector_type(4)))  int    v4i;

#define D_MODEL   1024
#define NUM_HEADS 16
#define D_HEAD    64
#define SEQ_T     2048
#define BATCH     2
#define M_TOT     (BATCH * SEQ_T)   // 4096
#define NEG_INF   (-1.0e9f)

// ---- CDNA5 async global->LDS path (compile-time probed) -------------------
#if defined(__AMDGCN__) && __has_builtin(__builtin_amdgcn_global_load_async_to_lds_b128)
#define HAS_ASYNC_LDS 1
#else
#define HAS_ASYNC_LDS 0
#endif

__device__ __forceinline__ void cp16_to_lds(__bf16* lds_dst, const __bf16* gsrc) {
#if HAS_ASYNC_LDS
    // builtin signature (from clang diagnostic): (int4 AS1*, int4 AS3*, imm, imm)
    __builtin_amdgcn_global_load_async_to_lds_b128(
        (__attribute__((address_space(1))) v4i*)gsrc,
        (__attribute__((address_space(3))) v4i*)lds_dst, 0, 0);
    __builtin_amdgcn_global_load_async_to_lds_b128(
        (__attribute__((address_space(1))) v4i*)(gsrc + 8),
        (__attribute__((address_space(3))) v4i*)(lds_dst + 8), 0, 0);
#else
    ((uint4*)lds_dst)[0] = ((const uint4*)gsrc)[0];
    ((uint4*)lds_dst)[1] = ((const uint4*)gsrc)[1];
#endif
}

__device__ __forceinline__ void async_wait_all() {
#if HAS_ASYNC_LDS
#if __has_builtin(__builtin_amdgcn_s_wait_asynccnt)
    __builtin_amdgcn_s_wait_asynccnt(0);
#else
    asm volatile("s_wait_asynccnt 0x0" ::: "memory");
#endif
#endif
}

union FragBF {
    v16bf v;
    uint4 q[2];
};

// A-fragment (16x32 MxK, bf16), per ISA 7.12.2:
//   lane l: m = l&15 ; hi = (l>>4)*8 ; holds K = [hi,hi+8) and [16+hi,16+hi+8)
__device__ __forceinline__ v16bf load_frag_a(const __bf16* base, int ld, int k0, int lane) {
    int m  = lane & 15;
    int hi = (lane >> 4) << 3;
    const __bf16* p = base + (size_t)m * ld + k0;
    FragBF f;
    f.q[0] = *(const uint4*)(p + hi);
    f.q[1] = *(const uint4*)(p + 16 + hi);
    return f.v;
}

// B-fragment (32x16 KxN, bf16); B stored transposed as [N][K] row-major:
//   lane l: n = l&15 ; kb = (l>>4)*8 ; holds K = [kb,kb+8) and [16+kb,16+kb+8)
__device__ __forceinline__ v16bf load_frag_b(const __bf16* base, int ld, int k0, int lane) {
    int n  = lane & 15;
    int kb = (lane >> 4) << 3;
    const __bf16* p = base + (size_t)n * ld + k0 + kb;
    FragBF f;
    f.q[0] = *(const uint4*)(p);
    f.q[1] = *(const uint4*)(p + 16);
    return f.v;
}

__device__ __forceinline__ v8f wmma_bf16(v16bf a, v16bf b, v8f c) {
    return __builtin_amdgcn_wmma_f32_16x16x32_bf16(
        /*neg_a=*/false, a, /*neg_b=*/false, b,
        /*c_mod=*/(short)0, c, /*reuse_a=*/false, /*reuse_b=*/false);
}

// ---------------------------------------------------------------------------
// fp32 -> bf16 conversion (vector x4)
// ---------------------------------------------------------------------------
__global__ void cvt_f32_to_bf16(const float* __restrict__ in,
                                __bf16* __restrict__ out, int n4) {
    int i = blockIdx.x * blockDim.x + threadIdx.x;
    if (i >= n4) return;
    float4 v = ((const float4*)in)[i];
    v4bf o;
    o.x = (__bf16)v.x; o.y = (__bf16)v.y; o.z = (__bf16)v.z; o.w = (__bf16)v.w;
    ((v4bf*)out)[i] = o;
}

// ---------------------------------------------------------------------------
// Tiled WMMA GEMM: C[m,n] = sum_k A[m,k]*W[n,k] + bias[n]
//   A: [4096 x 1024] bf16 row-major, W: [1024 x 1024] bf16 row-major ([N][K])
//   Block tile 128x128, 8 waves (4 along M, 2 along N), wave tile 32x64.
//   K staged in 64-wide double-buffered LDS tiles via async global->LDS.
// MODE 0: out bf16 at [b][h][t][d]   (Q, K)
// MODE 1: out bf16 at [b][h][d][t]   (V transposed)
// MODE 2: out f32  at [m][n]         (final output projection)
// ---------------------------------------------------------------------------
#define BK       64
#define LDSLD    72   // row stride in elements (144B, 16B aligned, conflict-spread)
#define NSTAGES  (D_MODEL / BK)   // 16

template <int MODE>
__global__ __launch_bounds__(256) void gemm_bf16_wmma(
    const __bf16* __restrict__ A,
    const __bf16* __restrict__ W,
    const float*  __restrict__ bias,
    void* __restrict__ Out) {
    __shared__ __bf16 sA[2][128 * LDSLD];
    __shared__ __bf16 sB[2][128 * LDSLD];

    const int K  = D_MODEL;
    const int m0 = blockIdx.y * 128;
    const int n0 = blockIdx.x * 128;
    const int tid  = threadIdx.x;
    const int lane = tid & 31;
    const int w    = tid >> 5;
    const int wm   = (w & 3) * 32;    // wave M offset within block tile
    const int wn   = (w >> 2) * 64;   // wave N offset within block tile

    // cooperative stage mapping: 128 rows x 4 segs of 16 bf16 (32B);
    // thread t handles rows (t>>2) and (t>>2)+64, segment (t&3)*16.
    const int lrow = tid >> 2;           // 0..63
    const int lseg = (tid & 3) * 16;     // 0,16,32,48

    const __bf16* Ag0 = A + (size_t)(m0 + lrow)      * K + lseg;
    const __bf16* Ag1 = A + (size_t)(m0 + lrow + 64) * K + lseg;
    const __bf16* Wg0 = W + (size_t)(n0 + lrow)      * K + lseg;
    const __bf16* Wg1 = W + (size_t)(n0 + lrow + 64) * K + lseg;

    v8f zero = {};
    v8f acc[2][4];
#pragma unroll
    for (int i = 0; i < 2; ++i)
#pragma unroll
        for (int j = 0; j < 4; ++j) acc[i][j] = zero;

    // stage issue: 8 async b128 per thread
    auto issue = [&](int buf, int kk) {
        cp16_to_lds(&sA[buf][lrow * LDSLD + lseg],        Ag0 + kk);
        cp16_to_lds(&sA[buf][(lrow + 64) * LDSLD + lseg], Ag1 + kk);
        cp16_to_lds(&sB[buf][lrow * LDSLD + lseg],        Wg0 + kk);
        cp16_to_lds(&sB[buf][(lrow + 64) * LDSLD + lseg], Wg1 + kk);
        __builtin_prefetch(Ag0 + kk + BK, 0, 1);   // hint next-next chunk toward L2
        __builtin_prefetch(Wg0 + kk + BK, 0, 1);
    };

    issue(0, 0);
    async_wait_all();
    __syncthreads();

    for (int s = 0; s < NSTAGES; ++s) {
        const int buf = s & 1;
        if (s + 1 < NSTAGES) issue(buf ^ 1, (s + 1) * BK);  // prefetch overlaps WMMAs

#pragma unroll
        for (int kc = 0; kc < BK; kc += 32) {
            v16bf a0 = load_frag_a(&sA[buf][(wm + 0)  * LDSLD], LDSLD, kc, lane);
            v16bf a1 = load_frag_a(&sA[buf][(wm + 16) * LDSLD], LDSLD, kc, lane);
            v16bf b0 = load_frag_b(&sB[buf][(wn + 0)  * LDSLD], LDSLD, kc, lane);
            v16bf b1 = load_frag_b(&sB[buf][(wn + 16) * LDSLD], LDSLD, kc, lane);
            v16bf b2 = load_frag_b(&sB[buf][(wn + 32) * LDSLD], LDSLD, kc, lane);
            v16bf b3 = load_frag_b(&sB[buf][(wn + 48) * LDSLD], LDSLD, kc, lane);

            acc[0][0] = wmma_bf16(a0, b0, acc[0][0]);
            acc[0][1] = wmma_bf16(a0, b1, acc[0][1]);
            acc[0][2] = wmma_bf16(a0, b2, acc[0][2]);
            acc[0][3] = wmma_bf16(a0, b3, acc[0][3]);
            acc[1][0] = wmma_bf16(a1, b0, acc[1][0]);
            acc[1][1] = wmma_bf16(a1, b1, acc[1][1]);
            acc[1][2] = wmma_bf16(a1, b2, acc[1][2]);
            acc[1][3] = wmma_bf16(a1, b3, acc[1][3]);
        }

        if (s + 1 < NSTAGES) {
            async_wait_all();   // next tile landed (my ops; barrier covers the rest)
            __syncthreads();    // also fences reads of buf before it is re-staged
        }
    }

    // Epilogue. C layout: lane 0-15 -> rows j, N=lane ; lane 16-31 -> rows j+8.
    const int col    = lane & 15;
    const int rhalf  = (lane >> 4) * 8;
#pragma unroll
    for (int mi = 0; mi < 2; ++mi) {
#pragma unroll
        for (int nt = 0; nt < 4; ++nt) {
            int n = n0 + wn + nt * 16 + col;
            float bv = bias[n];
#pragma unroll
            for (int j = 0; j < 8; ++j) {
                int m = m0 + wm + mi * 16 + rhalf + j;
                float val = acc[mi][nt][j] + bv;
                if (MODE == 2) {
                    ((float*)Out)[(size_t)m * D_MODEL + n] = val;
                } else {
                    int b = m >> 11, t = m & (SEQ_T - 1);
                    int h = n >> 6, d = n & (D_HEAD - 1);
                    size_t addr;
                    if (MODE == 0)
                        addr = ((size_t)((b * NUM_HEADS + h) * SEQ_T + t)) * D_HEAD + d;
                    else
                        addr = ((size_t)((b * NUM_HEADS + h) * D_HEAD + d)) * SEQ_T + t;
                    ((__bf16*)Out)[addr] = (__bf16)val;
                }
            }
        }
    }
}

// ---------------------------------------------------------------------------
// Flash attention: one wave (32 threads) per 16-query tile of one (b,h).
//   scores = Q K^T / 8, mask, online softmax, O += P V ; V is pre-transposed.
// Output written bf16 directly in [b][t][h*64+d] layout for the final GEMM.
// ---------------------------------------------------------------------------
__global__ __launch_bounds__(32) void flash_attn_wmma(
    const __bf16* __restrict__ Q,    // [B,H,T,dh]
    const __bf16* __restrict__ Km,   // [B,H,T,dh]
    const __bf16* __restrict__ Vt,   // [B,H,dh,T]
    const int*    __restrict__ mask, // [B,T]
    __bf16*       __restrict__ Omat) // [B*T, D]
{
    __shared__ __bf16 sP[16 * 40];   // 16x32 P tile, stride 40 for alignment

    const int lane = threadIdx.x;
    const int qt = blockIdx.x, h = blockIdx.y, b = blockIdx.z;

    const __bf16* Qh = Q  + ((size_t)(b * NUM_HEADS + h) * SEQ_T + qt * 16) * D_HEAD;
    const __bf16* Kh = Km + (size_t)(b * NUM_HEADS + h) * SEQ_T * D_HEAD;
    const __bf16* Vh = Vt + (size_t)(b * NUM_HEADS + h) * D_HEAD * SEQ_T;
    const int*    mb = mask + b * SEQ_T;

    // Q fragments for the two 32-wide d halves
    v16bf qa0 = load_frag_a(Qh, D_HEAD, 0,  lane);
    v16bf qa1 = load_frag_a(Qh, D_HEAD, 32, lane);

    v8f zero = {};
    v8f accO[4];
#pragma unroll
    for (int nt = 0; nt < 4; ++nt) accO[nt] = zero;
    float mrow[8], lrow[8];
#pragma unroll
    for (int j = 0; j < 8; ++j) { mrow[j] = -3.0e38f; lrow[j] = 0.0f; }

    const int col = lane & 15;
    const int rh  = lane >> 4;

    for (int kb = 0; kb < SEQ_T; kb += 32) {
        // ---- scores: two 16x16 tiles (keys kb..kb+15, kb+16..kb+31) ----
        v8f s0 = zero, s1 = zero;
        v16bf k00 = load_frag_b(Kh + (size_t)(kb)      * D_HEAD, D_HEAD, 0,  lane);
        v16bf k01 = load_frag_b(Kh + (size_t)(kb + 16) * D_HEAD, D_HEAD, 0,  lane);
        s0 = wmma_bf16(qa0, k00, s0);
        s1 = wmma_bf16(qa0, k01, s1);
        v16bf k10 = load_frag_b(Kh + (size_t)(kb)      * D_HEAD, D_HEAD, 32, lane);
        v16bf k11 = load_frag_b(Kh + (size_t)(kb + 16) * D_HEAD, D_HEAD, 32, lane);
        s0 = wmma_bf16(qa1, k10, s0);
        s1 = wmma_bf16(qa1, k11, s1);

        // ---- scale + mask (column = key index of this lane) ----
        bool valid0 = mb[kb + col] != 0;
        bool valid1 = mb[kb + 16 + col] != 0;
#pragma unroll
        for (int j = 0; j < 8; ++j) {
            s0[j] = valid0 ? s0[j] * 0.125f : NEG_INF;
            s1[j] = valid1 ? s1[j] * 0.125f : NEG_INF;
        }

        // ---- online softmax (row = fixed per VGPR j, spread across 16 lanes) ----
#pragma unroll
        for (int j = 0; j < 8; ++j) {
            float tm = fmaxf(s0[j], s1[j]);
            tm = fmaxf(tm, __shfl_xor(tm, 1, 16));
            tm = fmaxf(tm, __shfl_xor(tm, 2, 16));
            tm = fmaxf(tm, __shfl_xor(tm, 4, 16));
            tm = fmaxf(tm, __shfl_xor(tm, 8, 16));
            float mnew = fmaxf(mrow[j], tm);
            float fsc  = __expf(mrow[j] - mnew);
            float p0   = __expf(s0[j] - mnew);
            float p1   = __expf(s1[j] - mnew);
            float rs = p0 + p1;
            rs += __shfl_xor(rs, 1, 16);
            rs += __shfl_xor(rs, 2, 16);
            rs += __shfl_xor(rs, 4, 16);
            rs += __shfl_xor(rs, 8, 16);
            lrow[j] = lrow[j] * fsc + rs;
            mrow[j] = mnew;
#pragma unroll
            for (int nt = 0; nt < 4; ++nt) accO[nt][j] *= fsc;
            // stage P (C-layout -> row-major bf16) in LDS for A-fragment reload
            int r = j + 8 * rh;
            sP[r * 40 + col]      = (__bf16)p0;
            sP[r * 40 + 16 + col] = (__bf16)p1;
        }

        // ---- O += P @ V  (single wave: LDS ops are in-order) ----
        v16bf pa = load_frag_a(sP, 40, 0, lane);
#pragma unroll
        for (int nt = 0; nt < 4; ++nt) {
            v16bf vb = load_frag_b(Vh + (size_t)(nt * 16) * SEQ_T, SEQ_T, kb, lane);
            accO[nt] = wmma_bf16(pa, vb, accO[nt]);
        }
    }

    // ---- normalize and store bf16 into [b*T + t][h*64 + d] ----
#pragma unroll
    for (int j = 0; j < 8; ++j) {
        int r = j + 8 * rh;
        float inv = 1.0f / lrow[j];
        size_t m = (size_t)b * SEQ_T + qt * 16 + r;
#pragma unroll
        for (int nt = 0; nt < 4; ++nt) {
            int n = h * D_HEAD + nt * 16 + col;
            Omat[m * D_MODEL + n] = (__bf16)(accO[nt][j] * inv);
        }
    }
}

// ---------------------------------------------------------------------------
// Host-side orchestration
// ---------------------------------------------------------------------------
extern "C" void kernel_launch(void* const* d_in, const int* in_sizes, int n_in,
                              void* d_out, int out_size, void* d_ws, size_t ws_size,
                              hipStream_t stream) {
    (void)in_sizes; (void)n_in; (void)out_size; (void)ws_size;

    const float* x    = (const float*)d_in[0];
    const int*   amsk = (const int*)d_in[1];
    const float* Wq   = (const float*)d_in[2];
    const float* Wq_b = (const float*)d_in[3];
    const float* Wk   = (const float*)d_in[4];
    const float* Wk_b = (const float*)d_in[5];
    const float* Wv   = (const float*)d_in[6];
    const float* Wv_b = (const float*)d_in[7];
    const float* Wo   = (const float*)d_in[8];
    const float* Wo_b = (const float*)d_in[9];
    float* out = (float*)d_out;

    // workspace layout (bytes)
    const size_t SZ_X  = (size_t)M_TOT * D_MODEL * 2;     // 8 MB
    const size_t SZ_W  = (size_t)D_MODEL * D_MODEL * 2;   // 2 MB
    char* ws = (char*)d_ws;
    __bf16* xb   = (__bf16*)(ws);
    __bf16* wqb  = (__bf16*)(ws + SZ_X);
    __bf16* wkb  = (__bf16*)(ws + SZ_X + SZ_W);
    __bf16* wvb  = (__bf16*)(ws + SZ_X + 2 * SZ_W);
    __bf16* wob  = (__bf16*)(ws + SZ_X + 3 * SZ_W);
    __bf16* Qb   = (__bf16*)(ws + SZ_X + 4 * SZ_W);
    __bf16* Kb   = (__bf16*)(ws + 2 * SZ_X + 4 * SZ_W);
    __bf16* Vtb  = (__bf16*)(ws + 3 * SZ_X + 4 * SZ_W);
    __bf16* Omat = (__bf16*)(ws + 4 * SZ_X + 4 * SZ_W);   // total 48 MB

    // 1) fp32 -> bf16 conversions
    {
        int n4 = (M_TOT * D_MODEL) / 4;
        cvt_f32_to_bf16<<<(n4 + 255) / 256, 256, 0, stream>>>(x, xb, n4);
        int w4 = (D_MODEL * D_MODEL) / 4;
        cvt_f32_to_bf16<<<(w4 + 255) / 256, 256, 0, stream>>>(Wq, wqb, w4);
        cvt_f32_to_bf16<<<(w4 + 255) / 256, 256, 0, stream>>>(Wk, wkb, w4);
        cvt_f32_to_bf16<<<(w4 + 255) / 256, 256, 0, stream>>>(Wv, wvb, w4);
        cvt_f32_to_bf16<<<(w4 + 255) / 256, 256, 0, stream>>>(Wo, wob, w4);
    }

    // 2) QKV projections (Q,K in [b,h,t,d]; V transposed to [b,h,d,t])
    dim3 ggrid(D_MODEL / 128, M_TOT / 128);  // (8, 32)
    gemm_bf16_wmma<0><<<ggrid, 256, 0, stream>>>(xb, wqb, Wq_b, Qb);
    gemm_bf16_wmma<0><<<ggrid, 256, 0, stream>>>(xb, wkb, Wk_b, Kb);
    gemm_bf16_wmma<1><<<ggrid, 256, 0, stream>>>(xb, wvb, Wv_b, Vtb);

    // 3) flash attention
    dim3 fgrid(SEQ_T / 16, NUM_HEADS, BATCH); // (128, 16, 2)
    flash_attn_wmma<<<fgrid, 32, 0, stream>>>(Qb, Kb, Vtb, amsk, Omat);

    // 4) output projection -> d_out (fp32)
    gemm_bf16_wmma<2><<<ggrid, 256, 0, stream>>>(Omat, wob, Wo_b, out);
}
